// Qwen3Attention_45896020525087
// MI455X (gfx1250) — compile-verified, matrix operand order checked by reading, not run on hardware
//
#include <hip/hip_runtime.h>
#include <hip/hip_bf16.h>

// ---------------------------------------------------------------------------
// Qwen3 attention layer for MI455X (gfx1250, wave32, WMMA bf16 16x16x32,
// TDM / async global->LDS staging, ds_swizzle reductions)
// ---------------------------------------------------------------------------
#define H_   32
#define KV_  8
#define D_   128
#define HID_ 2560
#define B_   2
#define S_   2048
#define EPS_ 1e-6f
#define LOG_THETA_ 13.815510557964274f   // ln(1e6)

typedef __attribute__((ext_vector_type(16))) __bf16 bf16x16;
typedef __attribute__((ext_vector_type(8)))  float  f32x8;
typedef __attribute__((ext_vector_type(4)))  unsigned int u32x4;
typedef __attribute__((ext_vector_type(8)))  int   i32x8;
typedef __attribute__((ext_vector_type(4)))  int   i32x4;

#define DEV __device__ __forceinline__
#define AS1 __attribute__((address_space(1)))
#define AS3 __attribute__((address_space(3)))

// ---- feature selection for the LDS staging path ---------------------------
#if __has_builtin(__builtin_amdgcn_tensor_load_to_lds) && __has_builtin(__builtin_amdgcn_s_wait_tensorcnt)
#define USE_TDM 1
#else
#define USE_TDM 0
#endif
#if !USE_TDM && __has_builtin(__builtin_amdgcn_global_load_async_to_lds_b128)
#define USE_ASYNC 1
#else
#define USE_ASYNC 0
#endif

// K-dim placement inside a 16-bit A/B fragment: VGPR j holds the pair
// (k0, k0+1); lanes 0-15 (g=0) and 16-31 (g=1) are offset by 8.
DEV int frag_k0(int j, int g) {
    return ((j < 4) ? (2 * j) : (16 + 2 * (j - 4))) + 8 * g;
}

DEV f32x8 wmma_bf16(const bf16x16& a, const bf16x16& b, const f32x8& c) {
    return __builtin_amdgcn_wmma_f32_16x16x32_bf16(
        false, a, false, b, (short)0, c, false, false);
}

DEV f32x8 zero8() {
    f32x8 z;
#pragma unroll
    for (int i = 0; i < 8; ++i) z[i] = 0.0f;
    return z;
}

// Single-DS-op butterfly: ds_swizzle_b32, 32-lane group xor mode.
template <int MASK>
DEV float swz_xor(float x) {
    return __int_as_float(
        __builtin_amdgcn_ds_swizzle(__float_as_int(x), (MASK << 10) | 0x1F));
}

#if USE_TDM
// Issue one TDM descriptor: load a 32x128 bf16 tile (row stride 128 elems in
// memory) into LDS with +16B padding every 256B row -> LDS row stride 272B.
// 6-arg builtin form (clang-23 / therock-10.0 headers):
//   (uint32x4 g0, int32x8 g1, int32x4 g2, int32x4 g3, int32x8 g4, i32 cpol)
DEV void tdm_load_tile(unsigned lds_off, const __bf16* gtile) {
    const unsigned long long ga = (unsigned long long)gtile;
    u32x4 g0;
    g0[0] = 1u;                                      // count=1, user descriptor
    g0[1] = lds_off;                                 // lds_addr (bytes)
    g0[2] = (unsigned)(ga & 0xFFFFFFFFu);            // global_addr[31:0]
    g0[3] = (unsigned)((ga >> 32) & 0x01FFFFFFu)     // global_addr[56:32]
          | (2u << 30);                              // type = 2 (image)
    i32x8 g1;
    g1[0] = (int)((1u << 16)                         // data_size = 2B
                | (1u << 20)                         // pad_enable
                | (5u << 22)                         // pad_interval: 64 dwords
                | (3u << 25));                       // pad_amount:   4 dwords
    g1[1] = (int)(128u << 16);                       // tensor_dim0 = 128
    g1[2] = (int)((unsigned)S_ << 16);               // tensor_dim1 = 2048
    g1[3] = (int)(128u << 16);                       // tile_dim0 = 128
    g1[4] = 32;                                      // tile_dim1 = 32
    g1[5] = 128;                                     // tensor_dim0_stride = 128
    g1[6] = 0;
    g1[7] = 0;
    i32x4 gz4;
    gz4[0] = gz4[1] = gz4[2] = gz4[3] = 0;           // 2D tensor: groups 2/3 null
    i32x8 gz8;
#pragma unroll
    for (int i = 0; i < 8; ++i) gz8[i] = 0;
    __builtin_amdgcn_tensor_load_to_lds(g0, g1, gz4, gz4, gz8, 0);
}
#endif

#if USE_ASYNC
DEV void async_b128(const __bf16* g, const __bf16* l) {
    AS1 void* gp = (AS1 void*)(unsigned long long)g;
    AS3 void* lp = (AS3 void*)(unsigned)(unsigned long long)l;
    __builtin_amdgcn_global_load_async_to_lds_b128(gp, lp, 0, 0);
}
template <int N>
DEV void wait_async() {
#if __has_builtin(__builtin_amdgcn_s_wait_asynccnt)
    __builtin_amdgcn_s_wait_asynccnt((short)N);
#else
    asm volatile("s_wait_asynccnt %0" ::"i"(N) : "memory");
#endif
}
#endif

// ===========================================================================
// Kernel 1: fused QKV projection + per-head RMSNorm + RoPE, fp32 -> bf16.
// grid.x = (B*S)/16 row tiles, grid.y = H + 2*KV head-columns, block = 256.
// ===========================================================================
__global__ __launch_bounds__(256) void qkv_norm_rope_kernel(
    const float* __restrict__ X,
    const float* __restrict__ Wq,
    const float* __restrict__ Wk,
    const float* __restrict__ Wv,
    const float* __restrict__ qnw,
    const float* __restrict__ knw,
    __bf16* __restrict__ Qo,            // [B, H,  S, D]
    __bf16* __restrict__ Ko,            // [B, KV, S, D]
    __bf16* __restrict__ Vo)            // [B, KV, S, D]
{
    __shared__ float ssq[16][8];
    __shared__ float norminv[16];
    __shared__ float tile[16][D_ + 1];

    const int t    = threadIdx.x;
    const int wave = t >> 5;
    const int lane = t & 31;
    const int ln   = lane & 15;
    const int g    = lane >> 4;

    const int row0 = blockIdx.x * 16;
    const int hidx = blockIdx.y;

    const float* Wp;
    int N, colbase, mode;                 // mode: 0=q, 1=k, 2=v
    if (hidx < H_)            { Wp = Wq; N = H_  * D_; colbase = hidx * D_;              mode = 0; }
    else if (hidx < H_ + KV_) { Wp = Wk; N = KV_ * D_; colbase = (hidx - H_) * D_;       mode = 1; }
    else                      { Wp = Wv; N = KV_ * D_; colbase = (hidx - H_ - KV_) * D_; mode = 2; }

    const int col0 = colbase + wave * 16;
    const int arow = row0 + ln;

    f32x8 acc = zero8();
    for (int kb = 0; kb < HID_; kb += 32) {
        __builtin_prefetch(&X[arow * HID_ + kb + 128]);
        bf16x16 a, b;
#pragma unroll
        for (int j = 0; j < 8; ++j) {
            const int k0 = frag_k0(j, g);
            a[2 * j]     = (__bf16)X[arow * HID_ + kb + k0];
            a[2 * j + 1] = (__bf16)X[arow * HID_ + kb + k0 + 1];
            b[2 * j]     = (__bf16)Wp[(kb + k0) * N + col0 + ln];
            b[2 * j + 1] = (__bf16)Wp[(kb + k0 + 1) * N + col0 + ln];
        }
        acc = wmma_bf16(a, b, acc);
    }

    if (mode == 2) {                       // V: no norm / no rope, just relayout
        const int kvh = hidx - H_ - KV_;
#pragma unroll
        for (int r = 0; r < 8; ++r) {
            const int rowg = row0 + r + 8 * g;
            const int bb = rowg >> 11, ss = rowg & (S_ - 1);
            Vo[(((size_t)bb * KV_ + kvh) * S_ + ss) * D_ + wave * 16 + ln] = (__bf16)acc[r];
        }
        return;
    }

    // ---- per-row sum of squares (row m = r + 8g over 16 lanes x 8 waves)
    float ss[8];
#pragma unroll
    for (int r = 0; r < 8; ++r) ss[r] = acc[r] * acc[r];
#pragma unroll
    for (int r = 0; r < 8; ++r) {
        ss[r] += swz_xor<1>(ss[r]);
        ss[r] += swz_xor<2>(ss[r]);
        ss[r] += swz_xor<4>(ss[r]);
        ss[r] += swz_xor<8>(ss[r]);
    }
    if (ln == 0) {
#pragma unroll
        for (int r = 0; r < 8; ++r) ssq[r + 8 * g][wave] = ss[r];
    }
    __syncthreads();
    if (t < 16) {
        float tot = 0.f;
#pragma unroll
        for (int w2 = 0; w2 < 8; ++w2) tot += ssq[t][w2];
        norminv[t] = rsqrtf(tot * (1.0f / D_) + EPS_);
    }
    __syncthreads();

    const float* nw = (mode == 0) ? qnw : knw;
#pragma unroll
    for (int r = 0; r < 8; ++r) {
        const int m = r + 8 * g;
        const int d = wave * 16 + ln;
        tile[m][d] = acc[r] * norminv[m] * nw[d];
    }
    __syncthreads();

    // ---- RoPE + store (2048 elems / 256 threads)
#pragma unroll
    for (int i = 0; i < 8; ++i) {
        const int e = i * 256 + t;
        const int m = e >> 7, d = e & (D_ - 1);
        const int rowg = row0 + m;
        const int bb = rowg >> 11, ss2 = rowg & (S_ - 1);
        const float x = tile[m][d];
        const float partner = (d < 64) ? -tile[m][d + 64] : tile[m][d - 64];
        const int fi = d & 63;
        const float invf = __expf(-(float)fi * (LOG_THETA_ / 64.0f));
        const float ang = (float)ss2 * invf;
        float sv, cv;
        __sincosf(ang, &sv, &cv);
        const float o = x * cv + partner * sv;
        if (mode == 0)
            Qo[(((size_t)bb * H_ + hidx) * S_ + ss2) * D_ + d] = (__bf16)o;
        else
            Ko[(((size_t)bb * KV_ + (hidx - H_)) * S_ + ss2) * D_ + d] = (__bf16)o;
    }
}

// ===========================================================================
// Kernel 2: causal GQA flash attention, online softmax, bf16 WMMA.
// grid = (S/128, H, B); block = 256 (8 waves, wave w owns 16 query rows).
// K/V tiles (32x128 bf16) double-buffered in padded LDS, staged via the
// Tensor Data Mover (or async global->LDS, or a plain copy as fallback) so
// tile kt+1 streams in while tile kt feeds the matrix cores.
// ===========================================================================
__global__ __launch_bounds__(256) void attn_kernel(
    const __bf16* __restrict__ Q,       // [B, H,  S, D]
    const __bf16* __restrict__ K,       // [B, KV, S, D]
    const __bf16* __restrict__ V,       // [B, KV, S, D]
    __bf16* __restrict__ O)             // [B, S, H*D]
{
    __shared__ __align__(16) __bf16 sK[2][32][136];
    __shared__ __align__(16) __bf16 sV[2][32][136];

    const int t    = threadIdx.x;
    const int wave = t >> 5;
    const int lane = t & 31;
    const int ln   = lane & 15;
    const int g    = lane >> 4;

    const int qbase = blockIdx.x * 128;
    const int h     = blockIdx.y;
    const int b     = blockIdx.z;
    const int kvh   = h >> 2;            // G = 4
    const int qw0   = qbase + wave * 16;
    const int qmax  = qw0 + 15;

    const __bf16* Qp = Q + ((size_t)(b * H_ + h) * S_) * D_;
    const __bf16* Kp = K + ((size_t)(b * KV_ + kvh) * S_) * D_;
    const __bf16* Vp = V + ((size_t)(b * KV_ + kvh) * S_) * D_;

    // Resident Q^T B-fragments: 4 chunks over D=128 (lane = query column).
    bf16x16 qf[4];
    const int qcol = qw0 + ln;
#pragma unroll
    for (int c = 0; c < 4; ++c)
#pragma unroll
        for (int j = 0; j < 8; ++j) {
            const int d0 = 32 * c + frag_k0(j, g);
            qf[c][2 * j]     = Qp[qcol * D_ + d0];
            qf[c][2 * j + 1] = Qp[qcol * D_ + d0 + 1];
        }

    f32x8 acc[8];
#pragma unroll
    for (int dt = 0; dt < 8; ++dt) acc[dt] = zero8();
    float mrun = -1e30f, lrun = 0.0f;

    const int nkt = (qbase + 128) >> 5;          // 32-key tiles for this block
    const float sc = 0.08838834764831845f;       // 1/sqrt(128)

    // ---- stage tile kt2 into buffer bufi ----------------------------------
    auto do_stage = [&](int kt2, int bufi) {
        const int kb2 = kt2 * 32;
#if USE_TDM
        if (wave == 0) {
            tdm_load_tile((unsigned)(unsigned long long)&sK[bufi][0][0],
                          Kp + (size_t)kb2 * D_);
            tdm_load_tile((unsigned)(unsigned long long)&sV[bufi][0][0],
                          Vp + (size_t)kb2 * D_);
        }
#elif USE_ASYNC
#pragma unroll
        for (int p = 0; p < 2; ++p) {
            const int u = t + p * 256;
            const int row = u >> 4, c8 = (u & 15) * 8;
            async_b128(Kp + (size_t)(kb2 + row) * D_ + c8, &sK[bufi][row][c8]);
            async_b128(Vp + (size_t)(kb2 + row) * D_ + c8, &sV[bufi][row][c8]);
        }
#else
#pragma unroll
        for (int p = 0; p < 2; ++p) {
            const int u = t + p * 256;
            const int row = u >> 4, c8 = (u & 15) * 8;
            *(uint4*)&sK[bufi][row][c8] = *(const uint4*)(Kp + (size_t)(kb2 + row) * D_ + c8);
            *(uint4*)&sV[bufi][row][c8] = *(const uint4*)(Vp + (size_t)(kb2 + row) * D_ + c8);
        }
#endif
    };

    do_stage(0, 0);

    for (int kt = 0; kt < nkt; ++kt) {
        const int kb   = kt * 32;
        const int bufi = kt & 1;
        const bool more = (kt + 1) < nkt;
        if (more) do_stage(kt + 1, bufi ^ 1);      // prefetch next tile
        // retire the stage feeding this iteration, keep the prefetch in flight
#if USE_TDM
        if (wave == 0) {
            if (more) __builtin_amdgcn_s_wait_tensorcnt((short)2);
            else      __builtin_amdgcn_s_wait_tensorcnt((short)0);
        }
#elif USE_ASYNC
        if (more) wait_async<4>();
        else      wait_async<0>();
#endif
        __syncthreads();

        if (kb <= qmax) {                          // wave-uniform predicate
            // ---- S^T tiles: rows = keys, cols = queries
            f32x8 st[2];
#pragma unroll
            for (int tt = 0; tt < 2; ++tt) {
                st[tt] = zero8();
#pragma unroll
                for (int c = 0; c < 4; ++c) {
                    bf16x16 kf;
                    const int kr = tt * 16 + ln;
#pragma unroll
                    for (int j = 0; j < 8; ++j) {
                        const int d0 = 32 * c + frag_k0(j, g);
                        kf[2 * j]     = sK[bufi][kr][d0];
                        kf[2 * j + 1] = sK[bufi][kr][d0 + 1];
                    }
                    st[tt] = wmma_bf16(kf, qf[c], st[tt]);
                }
            }
            // ---- scale + causal mask + row max (lane owns query qw0+ln)
            const int qg_ = qw0 + ln;
            float tmax = -1e30f;
#pragma unroll
            for (int tt = 0; tt < 2; ++tt)
#pragma unroll
                for (int r = 0; r < 8; ++r) {
                    const int kgl = kb + tt * 16 + r + 8 * g;
                    const float v = (kgl <= qg_) ? st[tt][r] * sc : -1e30f;
                    st[tt][r] = v;
                    tmax = fmaxf(tmax, v);
                }
            tmax = fmaxf(tmax, swz_xor<16>(tmax));
            const float mnew  = fmaxf(mrun, tmax);
            const float alpha = __expf(mrun - mnew);
            float psum = 0.0f;
            bf16x16 pf;                            // P A-fragment, register-local
#pragma unroll
            for (int tt = 0; tt < 2; ++tt)
#pragma unroll
                for (int r = 0; r < 8; ++r) {
                    const float p = __expf(st[tt][r] - mnew);
                    psum += p;
                    pf[tt * 8 + r] = (__bf16)p;
                }
            psum += swz_xor<16>(psum);
            lrun = lrun * alpha + psum;
            mrun = mnew;
            // ---- rescale accumulators (alpha for row m fetched from lane m)
            float ar[8];
#pragma unroll
            for (int r = 0; r < 8; ++r) ar[r] = __shfl(alpha, r + 8 * g, 32);
#pragma unroll
            for (int dt = 0; dt < 8; ++dt)
#pragma unroll
                for (int r = 0; r < 8; ++r) acc[dt][r] *= ar[r];
            // ---- P @ V over the 8 d-tiles
#pragma unroll
            for (int dt = 0; dt < 8; ++dt) {
                bf16x16 vf;
#pragma unroll
                for (int j = 0; j < 8; ++j) {
                    const int k0 = frag_k0(j, g);
                    const int d  = dt * 16 + ln;
                    vf[2 * j]     = sV[bufi][k0][d];
                    vf[2 * j + 1] = sV[bufi][k0 + 1][d];
                }
                acc[dt] = wmma_bf16(pf, vf, acc[dt]);
            }
        }
        __syncthreads();                           // buffer reuse fence
    }

    // ---- finalize: divide by row sums, store bf16 in [B, S, H*D]
    float li[8];
#pragma unroll
    for (int r = 0; r < 8; ++r) li[r] = 1.0f / __shfl(lrun, r + 8 * g, 32);
#pragma unroll
    for (int dt = 0; dt < 8; ++dt)
#pragma unroll
        for (int r = 0; r < 8; ++r) {
            const int qglob = qw0 + r + 8 * g;
            const int d = dt * 16 + ln;
            O[((size_t)(b * S_ + qglob)) * (H_ * D_) + h * D_ + d] =
                (__bf16)(acc[dt][r] * li[r]);
        }
}

// ===========================================================================
// Kernel 3: output projection  out = O(bf16)[B*S, H*D] @ Wo(fp32)[H*D, HID]
// grid = ((B*S)/16, HID/128); block = 256 (8 waves x 16-col tiles).
// ===========================================================================
__global__ __launch_bounds__(256) void oproj_kernel(
    const __bf16* __restrict__ Oin,
    const float*  __restrict__ Wo,
    float* __restrict__ out)
{
    const int t    = threadIdx.x;
    const int wave = t >> 5;
    const int lane = t & 31;
    const int ln   = lane & 15;
    const int g    = lane >> 4;

    const int row0 = blockIdx.x * 16;
    const int col0 = blockIdx.y * 128 + wave * 16;
    const int arow = row0 + ln;

    f32x8 acc = zero8();
    for (int kb = 0; kb < H_ * D_; kb += 32) {
        __builtin_prefetch(&Oin[(size_t)arow * (H_ * D_) + kb + 256]);
        bf16x16 a, b;
#pragma unroll
        for (int j = 0; j < 8; ++j) {
            const int k0 = frag_k0(j, g);
            a[2 * j]     = Oin[(size_t)arow * (H_ * D_) + kb + k0];
            a[2 * j + 1] = Oin[(size_t)arow * (H_ * D_) + kb + k0 + 1];
            b[2 * j]     = (__bf16)Wo[(kb + k0) * HID_ + col0 + ln];
            b[2 * j + 1] = (__bf16)Wo[(kb + k0 + 1) * HID_ + col0 + ln];
        }
        acc = wmma_bf16(a, b, acc);
    }
#pragma unroll
    for (int r = 0; r < 8; ++r)
        out[(size_t)(row0 + r + 8 * g) * HID_ + col0 + ln] = acc[r];
}

// ===========================================================================
extern "C" void kernel_launch(void* const* d_in, const int* in_sizes, int n_in,
                              void* d_out, int out_size, void* d_ws, size_t ws_size,
                              hipStream_t stream) {
    const float* X   = (const float*)d_in[0];   // hidden_states [B,S,HID]
    // d_in[1] = position_ids (broadcast arange; positions derived analytically)
    const float* Wq  = (const float*)d_in[2];
    const float* Wk  = (const float*)d_in[3];
    const float* Wv  = (const float*)d_in[4];
    const float* Wo  = (const float*)d_in[5];
    const float* qnw = (const float*)d_in[6];
    const float* knw = (const float*)d_in[7];
    float* out = (float*)d_out;

    __bf16* wsb = (__bf16*)d_ws;
    __bf16* Qw  = wsb;                                    // B*H*S*D
    __bf16* Kw  = Qw + (size_t)B_ * H_  * S_ * D_;        // B*KV*S*D
    __bf16* Vw  = Kw + (size_t)B_ * KV_ * S_ * D_;
    __bf16* Ow  = Vw + (size_t)B_ * KV_ * S_ * D_;        // B*S*H*D

    dim3 g1((B_ * S_) / 16, H_ + 2 * KV_);
    qkv_norm_rope_kernel<<<g1, 256, 0, stream>>>(X, Wq, Wk, Wv, qnw, knw, Qw, Kw, Vw);

    dim3 g2(S_ / 128, H_, B_);
    attn_kernel<<<g2, 256, 0, stream>>>(Qw, Kw, Vw, Ow);

    dim3 g3((B_ * S_) / 16, HID_ / 128);
    oproj_kernel<<<g3, 256, 0, stream>>>(Ow, Wo, out);
}
